// MultiHeadAttention_73813307949347
// MI455X (gfx1250) — matrix operand order repaired and checked by reading, not compiled, not run on hardware
//
#include <hip/hip_runtime.h>

// ---------------------------------------------------------------------------
// Types for CDNA5 WMMA (wave32)
// ---------------------------------------------------------------------------
typedef __bf16 bf16_t;
typedef __attribute__((ext_vector_type(16))) __bf16 v16bf;
typedef __attribute__((ext_vector_type(8)))  __bf16 v8bf;
typedef __attribute__((ext_vector_type(8)))  float  v8f;

typedef __attribute__((ext_vector_type(4))) unsigned int u32x4;
typedef __attribute__((ext_vector_type(8))) int          i32x8;
typedef __attribute__((ext_vector_type(4))) int          i32x4;

#if defined(__has_builtin)
#if __has_builtin(__builtin_amdgcn_tensor_load_to_lds) && \
    __has_builtin(__builtin_amdgcn_s_wait_tensorcnt)
#define USE_TDM 1
#endif
#endif

#define NH_ 16
#define HD_ 128

static __device__ __forceinline__ v8f wmma_bf16(v16bf a, v16bf b, v8f c) {
  // D = A(16x32 bf16) * B(32x16 bf16) + C(16x16 f32)
  return __builtin_amdgcn_wmma_f32_16x16x32_bf16(false, a, false, b,
                                                 (short)0, c, false, false);
}

// A-matrix fragment (16x32, bf16), X row-major with leading dim ld.
// Layout: lane m=lane&15; lanes 0-15: halves 0..15 -> K=0..15,
// lanes 16-31: halves 0..15 -> K=16..31 (matches B fragments).
static __device__ __forceinline__ v16bf load_a_frag(const bf16_t* __restrict__ X,
                                                    int ld, int row, int k, int lane) {
  const bf16_t* p = X + (size_t)row * ld + k;
  const int kb = (lane < 16) ? 0 : 16;
  union { v16bf v; v8bf h[2]; } u;
  u.h[0] = *reinterpret_cast<const v8bf*>(p + kb);
  u.h[1] = *reinterpret_cast<const v8bf*>(p + kb + 8);
  return u.v;
}

// B-matrix fragment (32x16, bf16) from K-contiguous-per-column storage
// (i.e. W stored [N,K] row-major). Lanes 0-15: K=k+0..15, lanes 16-31: K=k+16..31.
static __device__ __forceinline__ v16bf load_b_frag(const bf16_t* __restrict__ W,
                                                    int ld, int col, int k, int lane) {
  const int kb = (lane < 16) ? 0 : 16;
  const bf16_t* p = W + (size_t)col * ld + k + kb;
  union { v16bf v; v8bf h[2]; } u;
  u.h[0] = *reinterpret_cast<const v8bf*>(p);
  u.h[1] = *reinterpret_cast<const v8bf*>(p + 8);
  return u.v;
}

// B-matrix fragment from an LDS row that is K-contiguous: row[k+kb .. k+kb+15].
static __device__ __forceinline__ v16bf load_b_frag_lds(const bf16_t* row, int k, int lane) {
  const int kb = (lane < 16) ? 0 : 16;
  union { v16bf v; v8bf h[2]; } u;
  u.h[0] = *reinterpret_cast<const v8bf*>(row + k + kb);
  u.h[1] = *reinterpret_cast<const v8bf*>(row + k + kb + 8);
  return u.v;
}

// ---------------------------------------------------------------------------
// fp32 -> bf16 conversion
// ---------------------------------------------------------------------------
__global__ void cvt_f32_bf16_kernel(const float* __restrict__ src,
                                    bf16_t* __restrict__ dst, size_t n) {
  size_t i = (size_t)blockIdx.x * blockDim.x + threadIdx.x;
  size_t stride = (size_t)gridDim.x * blockDim.x;
  for (; i < n; i += stride) dst[i] = (bf16_t)src[i];
}

// ---------------------------------------------------------------------------
// Y[M,N] = X[M,K] @ W[N,K]^T + bias[N]
// block = 128 threads (4 waves). Block tile 64x64, wave tile 16x64.
// ---------------------------------------------------------------------------
template <typename OutT>
__global__ void __launch_bounds__(128, 1)
gemm_bias_kernel(const bf16_t* __restrict__ X,
                 const bf16_t* __restrict__ W,
                 const float* __restrict__ bias,
                 OutT* __restrict__ Y,
                 int M, int N, int K) {
  const int lane = threadIdx.x & 31;
  const int wave = threadIdx.x >> 5;
  const int nBase = blockIdx.x * 64;
  const int mBase = blockIdx.y * 64 + wave * 16;

  v8f acc[4] = {};
  const int arow = mBase + (lane & 15);

  for (int k = 0; k < K; k += 32) {
    v16bf a = load_a_frag(X, K, arow, k, lane);
#pragma unroll
    for (int t = 0; t < 4; ++t) {
      v16bf b = load_b_frag(W, K, nBase + t * 16 + (lane & 15), k, lane);
      acc[t] = wmma_bf16(a, b, acc[t]);
    }
  }

  // C/D layout: lanes 0-15 rows 0..7, lanes 16-31 rows 8..15; col = lane&15.
  const int rOff = mBase + ((lane >> 4) << 3);
#pragma unroll
  for (int t = 0; t < 4; ++t) {
    const int col = nBase + t * 16 + (lane & 15);
    const float bv = bias ? bias[col] : 0.0f;
#pragma unroll
    for (int r = 0; r < 8; ++r) {
      float v = acc[t][r] + bv;
      Y[(size_t)(rOff + r) * N + col] = (OutT)v;
    }
  }
}

// ---------------------------------------------------------------------------
// Flash attention: one (b,h) per blockIdx.y, one 128-query tile per blockIdx.x.
// 256 threads = 8 waves x 16 query rows. K-tile staged into LDS by the Tensor
// Data Mover (TDM) with row padding; transposed V-tile staged by VALU path.
// Online softmax in f32; QK^T and P@V on WMMA. Row-sums of P via extra WMMA
// against an all-ones B fragment (FA2 style).
// ---------------------------------------------------------------------------
#define KT_PAD 136  // 128 + 8 halves  (row stride 272B -> distinct bank groups)
#define VT_PAD 40   // 32 + 8 halves   (row stride  80B -> distinct bank groups)

__global__ void __launch_bounds__(256, 1)
flash_attn_kernel(const bf16_t* __restrict__ Q,
                  const bf16_t* __restrict__ K,
                  const bf16_t* __restrict__ V,
                  bf16_t* __restrict__ O,
                  int S, int H, float scale) {
  __shared__ bf16_t Kt[32][KT_PAD];     // K-tile, row-major [key][d], padded rows
  __shared__ bf16_t Vt[HD_][VT_PAD];    // V-tile, transposed [d][key]
  __shared__ bf16_t Plds[8][16][32];    // per-wave P staging (C -> A layout)

  const int tid = threadIdx.x;
  const int lane = tid & 31;
  const int wave = tid >> 5;
  const int bh = blockIdx.y;
  const int b = bh / NH_;
  const int h = bh % NH_;

  const size_t headOff = (size_t)b * S * H + (size_t)h * HD_;
  const bf16_t* Qh = Q + headOff;
  const bf16_t* Kh = K + headOff;
  const bf16_t* Vh = V + headOff;
  bf16_t* Oh = O + headOff;

  const int qBase = blockIdx.x * 128 + wave * 16;

  // Per-thread staging assignment for V: 256 threads cover 32 keys x 128 d,
  // 16 halves (two 16B chunks) each.
  const int skr = tid >> 3;          // key row 0..31
  const int sdb = (tid & 7) * 16;    // d base 0..112

#ifdef USE_TDM
  // Constant part of the K-tile TDM descriptor (D# group 1):
  //   data_size=2B (code 1), pad_enable, pad_interval=64 DWORDs (row=256B),
  //   pad_amount=4 DWORDs (16B = 8 halves -> KT_PAD), tile 128 x 32,
  //   tensor dims H x S, dim0 stride H. workgroup_mask=0 (not in cluster).
  i32x8 tdmG1 = {};
  tdmG1[0] = (int)((1u << 16) | (1u << 20) | (5u << 22) | (3u << 25));
  tdmG1[1] = (int)((unsigned)H << 16);                          // tensor_dim0 lo16
  tdmG1[2] = (int)(((unsigned)H >> 16) | ((unsigned)S << 16));  // dim0 hi / dim1 lo
  tdmG1[3] = (int)(((unsigned)S >> 16) | (128u << 16));         // dim1 hi / tile_dim0
  tdmG1[4] = 32;                                                // tile_dim1 (rows)
  tdmG1[5] = H;                                                 // dim0 stride lo32
  const unsigned ldsKt = (unsigned)(size_t)(&Kt[0][0]);
#endif

  // All-ones B fragment: lsum += P @ ones puts rowsum(P) in every lane.
  v16bf ones;
#pragma unroll
  for (int e = 0; e < 16; ++e) ones[e] = (bf16_t)1.0f;

  // Preload Q fragments for this wave's 16 rows (HD=128 -> 4 chunks of 32).
  v16bf qf[4];
  const int qrow = qBase + (lane & 15);
#pragma unroll
  for (int i = 0; i < 4; ++i) qf[i] = load_a_frag(Qh, H, qrow, i * 32, lane);

  float m_i[8];
#pragma unroll
  for (int r = 0; r < 8; ++r) m_i[r] = -3.0e38f;
  v8f o[8] = {};   // 16 q-rows x 128 d, as 8 C-tiles (unnormalized)
  v8f lsum = {};   // running softmax denominator per row (same recurrence as o)

  const int cRow = (lane >> 4) << 3;  // row offset within C-layout
  const int cCol = lane & 15;

  for (int j = 0; j < S; j += 32) {
    __syncthreads();  // previous iteration's LDS reads complete

    // ---- stage K tile: TDM (async DMA, wave 0 issues) or VALU fallback
#ifdef USE_TDM
    if (wave == 0) {
      const unsigned long long ga =
          (unsigned long long)(size_t)(Kh + (size_t)j * H);
      u32x4 g0;
      g0[0] = 1u;                                   // count=1 (valid descriptor)
      g0[1] = ldsKt;                                // lds_addr
      g0[2] = (unsigned)(ga & 0xFFFFFFFFu);         // global_addr[31:0]
      g0[3] = ((unsigned)(ga >> 32) & 0x01FFFFFFu)  // global_addr[56:32]
              | (2u << 30);                         // type=2 ("image")
      i32x4 z4 = {};
      i32x8 z8 = {};
      __builtin_amdgcn_tensor_load_to_lds(g0, tdmG1, z4, z4, z8, 0);
    }
#else
    {
      const bf16_t* gk = Kh + (size_t)(j + skr) * H + sdb;
      v8bf k0 = *reinterpret_cast<const v8bf*>(gk);
      v8bf k1 = *reinterpret_cast<const v8bf*>(gk + 8);
      *reinterpret_cast<v8bf*>(&Kt[skr][sdb]) = k0;
      *reinterpret_cast<v8bf*>(&Kt[skr][sdb + 8]) = k1;
    }
#endif

    // ---- stage V tile (transposed) into LDS, overlapping the TDM copy
    {
      const bf16_t* gv = Vh + (size_t)(j + skr) * H + sdb;
      v8bf v0 = *reinterpret_cast<const v8bf*>(gv);
      v8bf v1 = *reinterpret_cast<const v8bf*>(gv + 8);
#pragma unroll
      for (int e = 0; e < 8; ++e) {
        Vt[sdb + e][skr] = v0[e];
        Vt[sdb + 8 + e][skr] = v1[e];
      }
    }

#ifdef USE_TDM
    if (wave == 0) __builtin_amdgcn_s_wait_tensorcnt(0);
#endif
    __syncthreads();  // staging visible to all waves

    // ---- scores: two 16x16 tiles over keys [j..j+15], [j+16..j+31]
    v8f s0 = {}, s1 = {};
    {
      const bf16_t* kr0 = &Kt[cCol][0];
      const bf16_t* kr1 = &Kt[16 + cCol][0];
#pragma unroll
      for (int i = 0; i < 4; ++i) {
        v16bf kb0 = load_b_frag_lds(kr0, i * 32, lane);
        s0 = wmma_bf16(qf[i], kb0, s0);
        v16bf kb1 = load_b_frag_lds(kr1, i * 32, lane);
        s1 = wmma_bf16(qf[i], kb1, s1);
      }
    }
#pragma unroll
    for (int r = 0; r < 8; ++r) { s0[r] *= scale; s1[r] *= scale; }

    // ---- row max: batched XOR-tree over 16-lane halves (8 rows per step,
    // so the 8 ds_bpermute ops per step can share one wait).
    float mx[8];
#pragma unroll
    for (int r = 0; r < 8; ++r) mx[r] = fmaxf(s0[r], s1[r]);
#pragma unroll
    for (int off = 1; off < 16; off <<= 1) {
      float t[8];
#pragma unroll
      for (int r = 0; r < 8; ++r) t[r] = __shfl_xor(mx[r], off, 32);
#pragma unroll
      for (int r = 0; r < 8; ++r) mx[r] = fmaxf(mx[r], t[r]);
    }

    // ---- online softmax state update
    float alpha[8], p0v[8], p1v[8];
#pragma unroll
    for (int r = 0; r < 8; ++r) {
      float mnew = fmaxf(m_i[r], mx[r]);
      alpha[r] = __expf(m_i[r] - mnew);
      p0v[r] = __expf(s0[r] - mnew);
      p1v[r] = __expf(s1[r] - mnew);
      m_i[r] = mnew;
    }

    // rescale running accumulators (O and l share the same recurrence)
#pragma unroll
    for (int dt = 0; dt < 8; ++dt)
#pragma unroll
      for (int r = 0; r < 8; ++r) o[dt][r] *= alpha[r];
#pragma unroll
    for (int r = 0; r < 8; ++r) lsum[r] *= alpha[r];

    // ---- P: C-layout -> LDS -> A-layout fragment (per-wave, LDS in-order)
#pragma unroll
    for (int r = 0; r < 8; ++r) {
      Plds[wave][cRow + r][cCol] = (bf16_t)p0v[r];
      Plds[wave][cRow + r][cCol + 16] = (bf16_t)p1v[r];
    }
    v16bf pf;
    {
      const int kb = (lane < 16) ? 0 : 16;
      const bf16_t* prow = &Plds[wave][lane & 15][0];
      union { v16bf v; v8bf h[2]; } u;
      u.h[0] = *reinterpret_cast<const v8bf*>(prow + kb);
      u.h[1] = *reinterpret_cast<const v8bf*>(prow + kb + 8);
      pf = u.v;
    }

    // ---- l += rowsum(P) via WMMA against ones (every lane gets the sum)
    lsum = wmma_bf16(pf, ones, lsum);

    // ---- O += P(16x32) @ V(32x128); V B-frags from transposed LDS tile
#pragma unroll
    for (int dt = 0; dt < 8; ++dt) {
      const int kb = (lane < 16) ? 0 : 16;
      const bf16_t* vr = &Vt[dt * 16 + cCol][kb];
      union { v16bf v; v8bf h[2]; } u;
      u.h[0] = *reinterpret_cast<const v8bf*>(vr);
      u.h[1] = *reinterpret_cast<const v8bf*>(vr + 8);
      o[dt] = wmma_bf16(pf, u.v, o[dt]);
    }
  }

  // ---- normalize and store
#pragma unroll
  for (int r = 0; r < 8; ++r) {
    const float inv = 1.0f / lsum[r];
    const size_t q = (size_t)(qBase + cRow + r);
#pragma unroll
    for (int dt = 0; dt < 8; ++dt) {
      Oh[q * H + dt * 16 + cCol] = (bf16_t)(o[dt][r] * inv);
    }
  }
}

// ---------------------------------------------------------------------------
// Host-side launch
// ---------------------------------------------------------------------------
extern "C" void kernel_launch(void* const* d_in, const int* in_sizes, int n_in,
                              void* d_out, int out_size, void* d_ws, size_t ws_size,
                              hipStream_t stream) {
  (void)in_sizes; (void)n_in; (void)out_size; (void)ws_size;

  const int B = 2, S = 2048, H = 2048;
  const int M = B * S;  // 4096
  const size_t nHS = (size_t)M * H;
  const size_t nW  = (size_t)H * H;

  const float* hs = (const float*)d_in[0];
  const float* Wq = (const float*)d_in[1];
  const float* bq = (const float*)d_in[2];
  const float* Wk = (const float*)d_in[3];
  const float* bk = (const float*)d_in[4];
  const float* Wv = (const float*)d_in[5];
  const float* bv = (const float*)d_in[6];
  const float* Wo = (const float*)d_in[7];
  const float* bo = (const float*)d_in[8];
  float* out = (float*)d_out;

  // workspace carve-out (256B-aligned slabs)
  size_t off = 0;
  auto take = [&](size_t elems) -> bf16_t* {
    bf16_t* p = reinterpret_cast<bf16_t*>((char*)d_ws + off);
    off += ((elems * sizeof(bf16_t) + 255) / 256) * 256;
    return p;
  };
  bf16_t* Xb  = take(nHS);
  bf16_t* Wqb = take(nW);
  bf16_t* Wkb = take(nW);
  bf16_t* Wvb = take(nW);
  bf16_t* Wob = take(nW);
  bf16_t* Qb  = take(nHS);
  bf16_t* Kb  = take(nHS);
  bf16_t* Vb  = take(nHS);
  bf16_t* Ob  = take(nHS);

  // 1) downcast activations + weights to bf16
  cvt_f32_bf16_kernel<<<4096, 256, 0, stream>>>(hs, Xb, nHS);
  cvt_f32_bf16_kernel<<<2048, 256, 0, stream>>>(Wq, Wqb, nW);
  cvt_f32_bf16_kernel<<<2048, 256, 0, stream>>>(Wk, Wkb, nW);
  cvt_f32_bf16_kernel<<<2048, 256, 0, stream>>>(Wv, Wvb, nW);
  cvt_f32_bf16_kernel<<<2048, 256, 0, stream>>>(Wo, Wob, nW);

  // 2) Q/K/V projections (bf16 out)
  dim3 gg(H / 64, M / 64);  // (32, 64)
  gemm_bias_kernel<bf16_t><<<gg, 128, 0, stream>>>(Xb, Wqb, bq, Qb, M, H, H);
  gemm_bias_kernel<bf16_t><<<gg, 128, 0, stream>>>(Xb, Wkb, bk, Kb, M, H, H);
  gemm_bias_kernel<bf16_t><<<gg, 128, 0, stream>>>(Xb, Wvb, bv, Vb, M, H, H);

  // 3) attention (flash, per-head), 128 query rows per block
  dim3 ga(S / 128, B * NH_);  // (16, 32)
  const float scale = 0.08838834764831845f;  // 1/sqrt(128)
  flash_attn_kernel<<<ga, 256, 0, stream>>>(Qb, Kb, Vb, Ob, S, H, scale);

  // 4) output projection (f32 out -> d_out)
  gemm_bias_kernel<float><<<gg, 128, 0, stream>>>(Ob, Wob, bo, out, M, H, H);
}
